// SuperModel_67104569033207
// MI455X (gfx1250) — matrix operand order repaired and checked by reading, not compiled, not run on hardware
//
#include <hip/hip_runtime.h>
#include <math.h>

// ---------------- problem constants ----------------
constexpr int BB  = 256;   // batch
constexpr int TT  = 256;   // time
constexpr int FF  = 40;    // raw features
constexpr int HH  = 256;   // hidden
constexpr int AAT = 64;    // attention dim
constexpr int HOR = 60;    // horizon
constexpr int DSP = 192;   // spatial dim
constexpr int G4  = 4 * HH;        // 1024 gate width
constexpr int KENC_CH = 14;        // (192+256)/32 K-chunks encoder
constexpr int KDEC_CH = 24;        // (512+256)/32 K-chunks decoder
constexpr int KPRJ_CH = 16;        // 512/32 K-chunks enc_proj

// ---------------- WMMA types ----------------
typedef __attribute__((ext_vector_type(16))) _Float16 v16h;
typedef __attribute__((ext_vector_type(8)))  float    v8f;

struct __attribute__((aligned(16))) U4 { unsigned int x, y, z, w; };

union Frag { v16h v; _Float16 h[16]; U4 q[2]; };

__device__ inline float sigf(float x) { return 1.0f / (1.0f + expf(-x)); }

// Load a WMMA A-fragment straight from a global row pointer.
// A order used here (contiguous k, matching the B swizzle): lane (m = lane&15, half = lane>>4) holds
//   elements 0..7  = A[m][16*half + 0..7]       (16 contiguous bytes)
//   elements 8..15 = A[m][16*half + 8..15]      (16 contiguous bytes)
__device__ inline void load_fragA(Frag& fa, const _Float16* rowp, int half) {
  fa.q[0] = *(const U4*)(rowp + 16 * half);
  fa.q[1] = *(const U4*)(rowp + 16 * half + 8);
}

__device__ inline void load_fragB(Frag& fb, const _Float16* tilep, int lane) {
  const U4* p = (const U4*)(tilep + lane * 16);
  fb.q[0] = p[0];
  fb.q[1] = p[1];
}

// =====================================================================
// zero init: encoder h-state (parity 0, both dirs) and encoder c-state
// =====================================================================
__global__ void k_zero(_Float16* __restrict__ h0, int nh, float* __restrict__ c0, int nc) {
  int i = blockIdx.x * blockDim.x + threadIdx.x;
  int stride = gridDim.x * blockDim.x;
  for (int j = i; j < nh; j += stride) h0[j] = (_Float16)0.0f;
  for (int j = i; j < nc; j += stride) c0[j] = 0.0f;
}

// =====================================================================
// weight swizzle into WMMA B-fragment order.
// tile (kc,nc): dst[lane*16+e] = W[kc*32 + 16*(lane>>4) + e][nc*16 + (lane&15)]
// =====================================================================
__global__ void k_swz_enc(const float* __restrict__ Wxf, const float* __restrict__ Whf,
                          const float* __restrict__ Wxb, const float* __restrict__ Whb,
                          _Float16* __restrict__ dst) {
  int i = blockIdx.x * 256 + threadIdx.x;
  if (i >= 2 * 64 * KENC_CH * 512) return;
  int le   = i & 511;
  int tile = i >> 9;
  int kc   = tile % KENC_CH;
  int tmp  = tile / KENC_CH;
  int nc   = tmp & 63;
  int dir  = tmp >> 6;
  int lane = le >> 4, e = le & 15;
  int k = kc * 32 + ((lane >> 4) << 4) + e;
  int n = nc * 16 + (lane & 15);
  const float* Wx = dir ? Wxb : Wxf;
  const float* Wh = dir ? Whb : Whf;
  float val = (k < DSP) ? Wx[k * G4 + n] : Wh[(k - DSP) * G4 + n];
  dst[i] = (_Float16)val;
}

__global__ void k_swz_dec(const float* __restrict__ Wxd, const float* __restrict__ Whd,
                          _Float16* __restrict__ dst) {
  int i = blockIdx.x * 256 + threadIdx.x;
  if (i >= 64 * KDEC_CH * 512) return;
  int le   = i & 511;
  int tile = i >> 9;
  int kc   = tile % KDEC_CH;
  int nc   = tile / KDEC_CH;
  int lane = le >> 4, e = le & 15;
  int k = kc * 32 + ((lane >> 4) << 4) + e;
  int n = nc * 16 + (lane & 15);
  // dec_in = [prev_y | context(512) | h(256)]; row 0 of Wx_d handled as rank-1 later
  float val = (k < 512) ? Wxd[(1 + k) * G4 + n] : Whd[(k - 512) * G4 + n];
  dst[i] = (_Float16)val;
}

__global__ void k_swz_prj(const float* __restrict__ We, _Float16* __restrict__ dst) {
  int i = blockIdx.x * 256 + threadIdx.x;
  if (i >= 4 * KPRJ_CH * 512) return;
  int le   = i & 511;
  int tile = i >> 9;
  int kc   = tile % KPRJ_CH;
  int nc   = tile / KPRJ_CH;
  int lane = le >> 4, e = le & 15;
  int k = kc * 32 + ((lane >> 4) << 4) + e;
  int n = nc * 16 + (lane & 15);
  dst[i] = (_Float16)We[k * AAT + n];
}

// =====================================================================
// spatial projection: hsp[t][b][192] = fp16(x[b,t,:40] @ W_sp + b_sp)
// one workgroup per (b,t), 192 threads
// =====================================================================
__global__ void k_spatial(const float* __restrict__ x, const float* __restrict__ Wsp,
                          const float* __restrict__ bsp, _Float16* __restrict__ hsp) {
  int bt = blockIdx.x;
  int b = bt / TT, t = bt % TT;
  __shared__ float xr[FF];
  if (threadIdx.x < FF) xr[threadIdx.x] = x[((size_t)b * TT + t) * FF + threadIdx.x];
  __syncthreads();
  int k = threadIdx.x;
  float acc = bsp[k];
#pragma unroll
  for (int f = 0; f < FF; ++f) acc += xr[f] * Wsp[f * DSP + k];
  hsp[((size_t)t * BB + b) * DSP + k] = (_Float16)acc;
}

// =====================================================================
// fused LSTM encoder step (both directions in one launch).
// grid: 512 blocks = dir(2) x jtile(16) x mtile(16); block: 128 threads = 4 waves.
// Wave g computes gate-g 16x16 tile of G = [x_t | h] @ [Wx; Wh].
// A-fragments loaded directly from global (no LDS staging, no K-loop barriers);
// gates meet in LDS once for the fused c/h update. h double-buffered by parity.
// =====================================================================
__global__ __launch_bounds__(128)
void k_enc_step(int s,
                const _Float16* __restrict__ hsp,     // [T][B][192]
                const _Float16* __restrict__ wsw,     // [2][64][14][512]
                const float* __restrict__ bf_, const float* __restrict__ bb_,
                _Float16* __restrict__ hstate,        // [2 parity][2 dir][B][H]
                float* __restrict__ cstate,           // [2 dir][B][H]
                _Float16* __restrict__ enc)           // [B][T][512]
{
  int bid   = blockIdx.x;
  int mtile = bid & 15;
  int jtile = (bid >> 4) & 15;
  int dir   = bid >> 8;
  int tid   = threadIdx.x;
  int wave  = tid >> 5;              // gate index 0..3
  int lane  = tid & 31;
  int half  = lane >> 4;
  int ts    = dir ? (TT - 1 - s) : s;

  __shared__ float Gsh[4][16][16];

  const _Float16* hin  = hstate + ((size_t)((s)     & 1) * 2 + dir) * (BB * HH);
  _Float16*       hout = hstate + ((size_t)((s + 1) & 1) * 2 + dir) * (BB * HH);

  int bA = mtile * 16 + (lane & 15);           // A row (batch index) for this lane
  const _Float16* arow_x = hsp + ((size_t)ts * BB + bA) * DSP;  // K 0..191
  const _Float16* arow_h = hin + (size_t)bA * HH;               // K 192..447

  v8f acc = {0, 0, 0, 0, 0, 0, 0, 0};
  int nc = (wave << 4) | jtile;      // gate*16 + jtile
  const _Float16* wtile = wsw + ((size_t)(dir * 64 + nc) * KENC_CH) * 512;

#pragma unroll
  for (int kc = 0; kc < KENC_CH; ++kc) {
    Frag fa, fb;
    const _Float16* rowp = (kc < 6) ? (arow_x + kc * 32) : (arow_h + (kc - 6) * 32);
    load_fragA(fa, rowp, half);
    load_fragB(fb, wtile + (size_t)kc * 512, lane);
    acc = __builtin_amdgcn_wmma_f32_16x16x32_f16(false, fa.v, false, fb.v,
                                                 (short)0, acc, false, false);
  }

  {   // C layout: acc[r] = C[r + 8*half][lane&15]
    int n = lane & 15;
#pragma unroll
    for (int r = 0; r < 8; ++r) Gsh[wave][r + 8 * half][n] = acc[r];
  }
  __syncthreads();

  const float* bias = dir ? bb_ : bf_;
  for (int q = tid; q < 256; q += 128) {
    int m = q >> 4, j = q & 15;
    int b  = mtile * 16 + m;
    int jg = jtile * 16 + j;
    float gi = Gsh[0][m][j] + bias[jg];
    float gf = Gsh[1][m][j] + bias[256 + jg];
    float gg = Gsh[2][m][j] + bias[512 + jg];
    float go = Gsh[3][m][j] + bias[768 + jg];
    size_t ci = (size_t)dir * BB * HH + (size_t)b * HH + jg;
    float c = sigf(gf) * cstate[ci] + sigf(gi) * tanhf(gg);
    cstate[ci] = c;
    float h = sigf(go) * tanhf(c);
    _Float16 hh = (_Float16)h;
    hout[(size_t)b * HH + jg] = hh;
    enc[((size_t)b * TT + ts) * 512 + dir * 256 + jg] = hh;
  }
}

// =====================================================================
// enc_proj = enc_out @ We : [65536,512] x [512,64], WMMA
// grid: 4096 M-tiles; block 128 = 4 waves (one N-tile each); no LDS needed.
// =====================================================================
__global__ __launch_bounds__(128)
void k_encproj(const _Float16* __restrict__ enc, const _Float16* __restrict__ wsw,
               float* __restrict__ proj) {
  int mtile = blockIdx.x;
  int tid = threadIdx.x, wave = tid >> 5, lane = tid & 31;
  int half = lane >> 4;
  v8f acc = {0, 0, 0, 0, 0, 0, 0, 0};
  const _Float16* wt = wsw + (size_t)wave * KPRJ_CH * 512;
  const _Float16* arow = enc + ((size_t)mtile * 16 + (lane & 15)) * 512;
#pragma unroll
  for (int kc = 0; kc < KPRJ_CH; ++kc) {
    Frag fa, fb;
    load_fragA(fa, arow + kc * 32, half);
    load_fragB(fb, wt + (size_t)kc * 512, lane);
    acc = __builtin_amdgcn_wmma_f32_16x16x32_f16(false, fa.v, false, fb.v,
                                                 (short)0, acc, false, false);
  }
  int n = lane & 15;
#pragma unroll
  for (int r = 0; r < 8; ++r) {
    size_t row = (size_t)mtile * 16 + r + 8 * half;
    proj[row * AAT + wave * 16 + n] = acc[r];
  }
}

// =====================================================================
// decoder init: enc_mean -> tanh projections; prev_y from x last row
// grid 256 (per batch row), block 256
// =====================================================================
__global__ void k_decinit(const _Float16* __restrict__ enc,
                          const float* __restrict__ Wih, const float* __restrict__ bih,
                          const float* __restrict__ Wic, const float* __restrict__ bic,
                          const float* __restrict__ x,
                          float* __restrict__ hdec, _Float16* __restrict__ hdecB,
                          float* __restrict__ cdec, float* __restrict__ py) {
  int b = blockIdx.x, tid = threadIdx.x;
  __shared__ float meanf[512];
  for (int f = tid; f < 512; f += 256) {
    float s = 0.0f;
    const _Float16* p = enc + ((size_t)b * TT) * 512 + f;
    for (int t = 0; t < TT; ++t) s += (float)p[(size_t)t * 512];
    meanf[f] = s * (1.0f / TT);
  }
  __syncthreads();
  int j = tid;
  float ah = bih[j], ac = bic[j];
  for (int f = 0; f < 512; ++f) {
    float m = meanf[f];
    ah += m * Wih[f * HH + j];
    ac += m * Wic[f * HH + j];
  }
  float h = tanhf(ah), c = tanhf(ac);
  hdec[b * HH + j]  = h;
  hdecB[b * HH + j] = (_Float16)h;   // parity 0
  cdec[b * HH + j]  = c;
  if (tid == 0)
    py[b] = 0.5f * (x[((size_t)b * TT + (TT - 1)) * FF + 0] +
                    x[((size_t)b * TT + (TT - 1)) * FF + 2]);
}

// =====================================================================
// attention: scores -> softmax over T -> context. One WG per batch row.
// enc (67.1MB fp16) + proj (16.8MB) stay L2-resident across the 60 steps.
// =====================================================================
__global__ void k_att(const float* __restrict__ hdec, const float* __restrict__ Wd,
                      const float* __restrict__ vvec, const float* __restrict__ proj,
                      const _Float16* __restrict__ enc, _Float16* __restrict__ ctxB) {
  int b = blockIdx.x, tid = threadIdx.x;
  __shared__ float hWd[AAT];
  __shared__ float sc[TT];
  __shared__ float red[TT];
  if (tid < AAT) {
    float a = 0.0f;
    for (int k = 0; k < HH; ++k) a += hdec[b * HH + k] * Wd[k * AAT + tid];
    hWd[tid] = a;
  }
  __syncthreads();
  {
    const float* pr = proj + ((size_t)b * TT + tid) * AAT;
    float s = 0.0f;
#pragma unroll 8
    for (int a = 0; a < AAT; ++a) s += tanhf(pr[a] + hWd[a]) * vvec[a];
    sc[tid] = s;
    red[tid] = s;
  }
  __syncthreads();
  for (int off = 128; off; off >>= 1) {
    if (tid < off) red[tid] = fmaxf(red[tid], red[tid + off]);
    __syncthreads();
  }
  float mx = red[0];
  __syncthreads();
  float e = expf(sc[tid] - mx);
  sc[tid]  = e;
  red[tid] = e;
  __syncthreads();
  for (int off = 128; off; off >>= 1) {
    if (tid < off) red[tid] += red[tid + off];
    __syncthreads();
  }
  float inv = 1.0f / red[0];
  sc[tid] *= inv;
  __syncthreads();
  for (int j = tid; j < 512; j += 256) {
    float acc = 0.0f;
    const _Float16* p = enc + ((size_t)b * TT) * 512 + j;
    for (int t = 0; t < TT; ++t) acc += sc[t] * (float)p[(size_t)t * 512];
    ctxB[b * 512 + j] = (_Float16)acc;
  }
}

// =====================================================================
// decoder LSTM cell, fused like the encoder step. A = [context | h],
// prev_y handled as rank-1 (py[b] * Wx_d row 0) in the gate epilogue.
// =====================================================================
__global__ __launch_bounds__(128)
void k_dec_cell(int d,
                const _Float16* __restrict__ ctxB,   // [B][512]
                const _Float16* __restrict__ wsw,    // [64][24][512]
                const float* __restrict__ bd, const float* __restrict__ Wxd0,
                const float* __restrict__ py,
                _Float16* __restrict__ hbf,          // [2][B][H]
                float* __restrict__ hdec, float* __restrict__ cdec) {
  int mtile = blockIdx.x & 15;
  int jtile = blockIdx.x >> 4;
  int tid = threadIdx.x, wave = tid >> 5, lane = tid & 31;
  int half = lane >> 4;

  __shared__ float Gsh[4][16][16];

  const _Float16* hin  = hbf + (size_t)((d)     & 1) * BB * HH;
  _Float16*       hout = hbf + (size_t)((d + 1) & 1) * BB * HH;

  int bA = mtile * 16 + (lane & 15);
  const _Float16* arow_c = ctxB + (size_t)bA * 512;   // K 0..511
  const _Float16* arow_h = hin + (size_t)bA * HH;     // K 512..767

  v8f acc = {0, 0, 0, 0, 0, 0, 0, 0};
  int nc = (wave << 4) | jtile;
  const _Float16* wtile = wsw + ((size_t)nc * KDEC_CH) * 512;

#pragma unroll
  for (int kc = 0; kc < KDEC_CH; ++kc) {
    Frag fa, fb;
    const _Float16* rowp = (kc < 16) ? (arow_c + kc * 32) : (arow_h + (kc - 16) * 32);
    load_fragA(fa, rowp, half);
    load_fragB(fb, wtile + (size_t)kc * 512, lane);
    acc = __builtin_amdgcn_wmma_f32_16x16x32_f16(false, fa.v, false, fb.v,
                                                 (short)0, acc, false, false);
  }

  {
    int n = lane & 15;
#pragma unroll
    for (int r = 0; r < 8; ++r) Gsh[wave][r + 8 * half][n] = acc[r];
  }
  __syncthreads();

  for (int q = tid; q < 256; q += 128) {
    int m = q >> 4, j = q & 15;
    int b  = mtile * 16 + m;
    int jg = jtile * 16 + j;
    float pyb = py[b];
    float gi = Gsh[0][m][j] + bd[jg]       + pyb * Wxd0[jg];
    float gf = Gsh[1][m][j] + bd[256 + jg] + pyb * Wxd0[256 + jg];
    float gg = Gsh[2][m][j] + bd[512 + jg] + pyb * Wxd0[512 + jg];
    float go = Gsh[3][m][j] + bd[768 + jg] + pyb * Wxd0[768 + jg];
    size_t ci = (size_t)b * HH + jg;
    float c = sigf(gf) * cdec[ci] + sigf(gi) * tanhf(gg);
    cdec[ci] = c;
    float h = sigf(go) * tanhf(c);
    hdec[ci] = h;
    hout[ci] = (_Float16)h;
  }
}

// =====================================================================
// y = h @ Wo + bo ; write d_out[:, d] and feed back prev_y
// =====================================================================
__global__ void k_out(int d, const float* __restrict__ hdec, const float* __restrict__ Wo,
                      const float* __restrict__ bo, float* __restrict__ out,
                      float* __restrict__ py) {
  int b = threadIdx.x;
  float a = bo[0];
  for (int j = 0; j < HH; ++j) a += hdec[b * HH + j] * Wo[j];
  out[b * HOR + d] = a;
  py[b] = a;
}

// =====================================================================
// host launcher
// =====================================================================
extern "C" void kernel_launch(void* const* d_in, const int* in_sizes, int n_in,
                              void* d_out, int out_size, void* d_ws, size_t ws_size,
                              hipStream_t stream) {
  (void)in_sizes; (void)n_in; (void)out_size; (void)ws_size;
  const float* x     = (const float*)d_in[0];
  const float* W_sp  = (const float*)d_in[1];
  const float* b_sp  = (const float*)d_in[2];
  const float* Wx_ef = (const float*)d_in[3];
  const float* Wh_ef = (const float*)d_in[4];
  const float* b_ef  = (const float*)d_in[5];
  const float* Wx_eb = (const float*)d_in[6];
  const float* Wh_eb = (const float*)d_in[7];
  const float* b_eb  = (const float*)d_in[8];
  const float* We    = (const float*)d_in[9];
  const float* Wd    = (const float*)d_in[10];
  const float* v     = (const float*)d_in[11];
  const float* Wx_d  = (const float*)d_in[12];
  const float* Wh_d  = (const float*)d_in[13];
  const float* b_d   = (const float*)d_in[14];
  const float* W_ih  = (const float*)d_in[15];
  const float* b_ih  = (const float*)d_in[16];
  const float* W_ic  = (const float*)d_in[17];
  const float* b_ic  = (const float*)d_in[18];
  const float* Wo    = (const float*)d_in[19];
  const float* bo    = (const float*)d_in[20];
  float* out = (float*)d_out;

  char* ws = (char*)d_ws;
  size_t off = 0;
  auto alloc = [&](size_t bytes) {
    size_t o = off;
    off += (bytes + 255) & ~(size_t)255;
    return o;
  };
  _Float16* hsp   = (_Float16*)(ws + alloc((size_t)TT * BB * DSP * 2));
  _Float16* encb  = (_Float16*)(ws + alloc((size_t)BB * TT * 512 * 2));
  float*    proj  = (float*)   (ws + alloc((size_t)BB * TT * AAT * 4));
  _Float16* wswE  = (_Float16*)(ws + alloc((size_t)2 * 64 * KENC_CH * 512 * 2));
  _Float16* wswD  = (_Float16*)(ws + alloc((size_t)64 * KDEC_CH * 512 * 2));
  _Float16* wswP  = (_Float16*)(ws + alloc((size_t)4 * KPRJ_CH * 512 * 2));
  _Float16* hEnc  = (_Float16*)(ws + alloc((size_t)2 * 2 * BB * HH * 2));
  float*    cEnc  = (float*)   (ws + alloc((size_t)2 * BB * HH * 4));
  float*    hDec  = (float*)   (ws + alloc((size_t)BB * HH * 4));
  _Float16* hDecB = (_Float16*)(ws + alloc((size_t)2 * BB * HH * 2));
  float*    cDec  = (float*)   (ws + alloc((size_t)BB * HH * 4));
  _Float16* ctxB  = (_Float16*)(ws + alloc((size_t)BB * 512 * 2));
  float*    py    = (float*)   (ws + alloc((size_t)BB * 4));

  k_zero<<<512, 256, 0, stream>>>(hEnc, 2 * BB * HH, cEnc, 2 * BB * HH);
  k_swz_enc<<<(2 * 64 * KENC_CH * 512 + 255) / 256, 256, 0, stream>>>(Wx_ef, Wh_ef, Wx_eb, Wh_eb, wswE);
  k_swz_dec<<<(64 * KDEC_CH * 512 + 255) / 256, 256, 0, stream>>>(Wx_d, Wh_d, wswD);
  k_swz_prj<<<(4 * KPRJ_CH * 512 + 255) / 256, 256, 0, stream>>>(We, wswP);
  k_spatial<<<BB * TT, DSP, 0, stream>>>(x, W_sp, b_sp, hsp);

  for (int s = 0; s < TT; ++s)
    k_enc_step<<<512, 128, 0, stream>>>(s, hsp, wswE, b_ef, b_eb, hEnc, cEnc, encb);

  k_encproj<<<(BB * TT) / 16, 128, 0, stream>>>(encb, wswP, proj);
  k_decinit<<<BB, 256, 0, stream>>>(encb, W_ih, b_ih, W_ic, b_ic, x, hDec, hDecB, cDec, py);

  for (int d = 0; d < HOR; ++d) {
    k_att<<<BB, 256, 0, stream>>>(hDec, Wd, v, proj, encb, ctxB);
    k_dec_cell<<<256, 128, 0, stream>>>(d, ctxB, wswD, b_d, Wx_d, py, hDecB, hDec, cDec);
    k_out<<<1, 256, 0, stream>>>(d, hDec, Wo, bo, out, py);
  }
}
